// GraniteSpeechConformerBlock_24472723653145
// MI455X (gfx1250) — matrix operand order repaired and checked by reading, not compiled, not run on hardware
//
#include <hip/hip_runtime.h>
#include <stdint.h>

// ---------------------------------------------------------------------------
// Types / helpers
// ---------------------------------------------------------------------------
typedef __attribute__((ext_vector_type(16))) __bf16          bf16x16;
typedef __attribute__((ext_vector_type(16))) unsigned short  u16x16;
typedef __attribute__((ext_vector_type(8)))  unsigned short  u16x8;
typedef __attribute__((ext_vector_type(8)))  float           f32x8;
typedef __attribute__((ext_vector_type(4)))  float           f32x4;

#if defined(__has_builtin)
#if __has_builtin(__builtin_amdgcn_global_load_async_to_lds_b128)
#define HAVE_ASYNC_LDS 1
#endif
#endif
#ifndef HAVE_ASYNC_LDS
#define HAVE_ASYNC_LDS 0
#endif

// Builtin expects int4-vector pointers: param 1 is
//   "__attribute__((__vector_size__(4*sizeof(int)))) int __device__ *"  (AS1)
typedef int v4i __attribute__((vector_size(16)));
typedef __attribute__((address_space(1))) v4i* gv4i_p;   // global src
typedef __attribute__((address_space(3))) v4i* lv4i_p;   // LDS dst

__device__ __forceinline__ unsigned short f2bf(float f) {
  unsigned u = __builtin_bit_cast(unsigned, f);
  u += 0x7fffu + ((u >> 16) & 1u);            // round-to-nearest-even
  return (unsigned short)(u >> 16);
}
__device__ __forceinline__ float bf2f(unsigned short s) {
  return __builtin_bit_cast(float, ((unsigned)s) << 16);
}
__device__ __forceinline__ bf16x16 pack_bf16(u16x8 lo, u16x8 hi) {
  u16x16 t;
#pragma unroll
  for (int i = 0; i < 8; ++i) { t[i] = lo[i]; t[8 + i] = hi[i]; }
  return __builtin_bit_cast(bf16x16, t);
}
__device__ __forceinline__ float siluf(float v) { return v / (1.f + __expf(-v)); }

// 16B global -> LDS copy.  Async (ASYNCcnt-tracked) when the toolchain exposes
// the gfx1250 builtin; otherwise plain staged copy (identical data movement).
__device__ __forceinline__ void cp_b128(unsigned short* lds_dst,
                                        const unsigned short* gsrc) {
#if HAVE_ASYNC_LDS
  __builtin_amdgcn_global_load_async_to_lds_b128(
      (gv4i_p)(uintptr_t)gsrc, (lv4i_p)(uint32_t)(uintptr_t)lds_dst, 0, 0);
#else
  *(u16x8*)lds_dst = *(const u16x8*)gsrc;
#endif
}
__device__ __forceinline__ void cp_wait() {
#if HAVE_ASYNC_LDS
#if __has_builtin(__builtin_amdgcn_s_wait_asynccnt)
  __builtin_amdgcn_s_wait_asynccnt(0);
#else
  asm volatile("s_wait_asynccnt 0" ::: "memory");
#endif
#endif
}

// ---------------------------------------------------------------------------
// LayerNorm: one wave (32 lanes) per row of D=1024.  OUT: 0 -> bf16, 1 -> f32
// ---------------------------------------------------------------------------
template <int OUT>
__global__ void __launch_bounds__(256) ln_kernel(
    const float* __restrict__ x, const float* __restrict__ g,
    const float* __restrict__ b, unsigned short* __restrict__ ob,
    float* __restrict__ of, int M, int D) {
  const int lane = threadIdx.x & 31;
  const int w    = threadIdx.x >> 5;
  const int row  = blockIdx.x * 8 + w;
  if (row >= M) return;
  const float* xr = x + (size_t)row * D;
  float s = 0.f, s2 = 0.f;
  for (int c = lane; c < D; c += 32) { float v = xr[c]; s += v; s2 += v * v; }
#pragma unroll
  for (int m = 16; m >= 1; m >>= 1) {
    s  += __shfl_xor(s,  m, 32);
    s2 += __shfl_xor(s2, m, 32);
  }
  const float mu  = s / (float)D;
  const float var = s2 / (float)D - mu * mu;
  const float rs  = rsqrtf(var + 1e-5f);
  for (int c = lane; c < D; c += 32) {
    float v = (xr[c] - mu) * rs * g[c] + b[c];
    if (OUT == 0) ob[(size_t)row * D + c] = f2bf(v);
    else          of[(size_t)row * D + c] = v;
  }
}

// ---------------------------------------------------------------------------
// bf16 WMMA GEMM: C = A(bf16)[M,K] * B(f32 weights)[K,N]  (+epilogue)
//   block tile 128x128, K-step 64, 256 threads = 8 waves, wave tile 32x64.
//   Pipeline: A double-buffered in LDS via async global->LDS copies issued for
//   tile t+1 during tile t's WMMAs; B (needs f32->bf16 convert) prefetched to
//   registers during compute, stored to LDS after the barrier.
// EPI: 0 = bias -> bf16 | 1 = silu(bias) -> bf16
//      2 = 0.5*(y+bias)+res -> f32 | 3 = (y+bias)+res -> f32
// M,N multiples of 128; K multiple of 64 (true for every call here)
// ---------------------------------------------------------------------------
#define GTM 128
#define GTN 128
#define GTK 64

template <int EPI>
__global__ void __launch_bounds__(256) gemm_bf16(
    const unsigned short* __restrict__ A, const float* __restrict__ B,
    const float* __restrict__ bias, const float* __restrict__ res,
    float* __restrict__ outF, unsigned short* __restrict__ outB,
    int M, int N, int K) {
  __shared__ __align__(16) unsigned short As[2][GTM][GTK + 8];
  __shared__ __align__(16) unsigned short Bs[GTN][GTK + 8];  // transposed [n][k]

  const int tid  = threadIdx.x;
  const int lane = tid & 31;
  const int wid  = tid >> 5;
  const int wm   = wid >> 1;            // 0..3  (row group of 32)
  const int wn   = wid & 1;             // 0..1  (col group of 64)
  const int m0   = blockIdx.x * GTM;
  const int n0   = blockIdx.y * GTN;
  const int lh   = lane >> 4;           // lane half
  const int lr   = lane & 15;
  const int ar   = tid >> 3;            // A-stage row  (0..31)
  const int av   = tid & 7;             // A-stage vec col

  f32x8 acc[2][4];
#pragma unroll
  for (int mi = 0; mi < 2; ++mi)
#pragma unroll
    for (int ni = 0; ni < 4; ++ni) acc[mi][ni] = (f32x8)0.0f;

  const int nT = K / GTK;
  f32x4 breg[8];

  // ---- prologue: tile 0 ----------------------------------------------------
#pragma unroll
  for (int i = 0; i < 4; ++i) {
    const int row = ar + i * 32;
    cp_b128(&As[0][row][av * 8], A + (size_t)(m0 + row) * K + av * 8);
  }
#pragma unroll
  for (int i = 0; i < 8; ++i) {
    const int idx = tid + i * 256;
    breg[i] = *(const f32x4*)(B + (size_t)(idx >> 5) * N + n0 + (idx & 31) * 4);
  }
  cp_wait();
#pragma unroll
  for (int i = 0; i < 8; ++i) {
    const int idx = tid + i * 256;
    const int kk = idx >> 5, nv = idx & 31;
#pragma unroll
    for (int j = 0; j < 4; ++j) Bs[nv * 4 + j][kk] = f2bf(breg[i][j]);
  }
  __syncthreads();

  for (int t = 0; t < nT; ++t) {
    const int  cur  = t & 1;
    const bool more = (t + 1) < nT;
    if (more) {
      const int k1 = (t + 1) * GTK;
#pragma unroll
      for (int i = 0; i < 4; ++i) {     // async fill of the other A buffer
        const int row = ar + i * 32;
        cp_b128(&As[cur ^ 1][row][av * 8],
                A + (size_t)(m0 + row) * K + k1 + av * 8);
      }
#pragma unroll
      for (int i = 0; i < 8; ++i) {     // B tile t+1 -> registers
        const int idx = tid + i * 256;
        breg[i] = *(const f32x4*)(B + (size_t)(k1 + (idx >> 5)) * N +
                                  n0 + (idx & 31) * 4);
      }
      if (t + 2 < nT)
        __builtin_prefetch(B + (size_t)((t + 2) * GTK + (tid >> 2)) * N +
                           n0 + (tid & 3) * 32, 0, 1);
    }

    // ---- compute on As[cur], Bs ---------------------------------------------
#pragma unroll
    for (int kk = 0; kk < GTK; kk += 32) {
      bf16x16 af[2], bfr[4];
#pragma unroll
      for (int mi = 0; mi < 2; ++mi) {
        // A 16x32 frag: lane<16 row=lr ks {0..7,16..23}; lane>=16 ks {8..15,24..31}
        const unsigned short* p = &As[cur][wm * 32 + mi * 16 + lr][kk + lh * 8];
        af[mi] = pack_bf16(*(const u16x8*)p, *(const u16x8*)(p + 16));
      }
#pragma unroll
      for (int ni = 0; ni < 4; ++ni) {
        // B 32x16 frag: lane col = lr, lane<16 ks {0..7,16..23}; lane>=16 ks {8..15,24..31}
        const unsigned short* p = &Bs[wn * 64 + ni * 16 + lr][kk + lh * 8];
        bfr[ni] = pack_bf16(*(const u16x8*)p, *(const u16x8*)(p + 16));
      }
#pragma unroll
      for (int mi = 0; mi < 2; ++mi)
#pragma unroll
        for (int ni = 0; ni < 4; ++ni)
          acc[mi][ni] = __builtin_amdgcn_wmma_f32_16x16x32_bf16(
              false, af[mi], false, bfr[ni], (short)0, acc[mi][ni], false, false);
    }
    __syncthreads();                    // all waves done reading Bs / As[cur]

    if (more) {
#pragma unroll
      for (int i = 0; i < 8; ++i) {     // commit B tile t+1 to LDS
        const int idx = tid + i * 256;
        const int kk = idx >> 5, nv = idx & 31;
#pragma unroll
        for (int j = 0; j < 4; ++j) Bs[nv * 4 + j][kk] = f2bf(breg[i][j]);
      }
      cp_wait();                        // A tile t+1 landed in LDS
    }
    __syncthreads();
  }

  // ---- epilogue: C layout: vgpr r -> row lh*8+r, col lr ---------------------
#pragma unroll
  for (int mi = 0; mi < 2; ++mi) {
#pragma unroll
    for (int ni = 0; ni < 4; ++ni) {
      const int col  = n0 + wn * 64 + ni * 16 + lr;
      const int rowb = m0 + wm * 32 + mi * 16 + lh * 8;
      const float bv = bias ? bias[col] : 0.f;
#pragma unroll
      for (int r = 0; r < 8; ++r) {
        const size_t o = (size_t)(rowb + r) * N + col;
        float v = acc[mi][ni][r] + bv;
        if (EPI == 0)      outB[o] = f2bf(v);
        else if (EPI == 1) outB[o] = f2bf(siluf(v));
        else if (EPI == 2) outF[o] = 0.5f * v + res[o];
        else               outF[o] = v + res[o];
      }
    }
  }
}

// ---------------------------------------------------------------------------
// Attention: one block (128 thr = 4 waves) per (b*nb*H, 16-query-row tile).
// C=200 context, DH=128, scores padded to 208 (13 j-tiles), probs to 224.
// ---------------------------------------------------------------------------
__global__ void __launch_bounds__(128) attn_kernel(
    const unsigned short* __restrict__ Q,    // [16000,1024] bf16
    const unsigned short* __restrict__ KV,   // [16000,2048] bf16 (k|v)
    const int* __restrict__ dist,            // [200,200]
    const float* __restrict__ rel,           // [1025,128]
    unsigned short* __restrict__ O) {        // [16000,1024] bf16
  __shared__ __align__(16) unsigned short Qs[16][136];
  __shared__ __align__(16) float          Ss[16][224];
  __shared__ __align__(16) unsigned short Vt[128][136];  // V chunk, [dim][j]

  const int tid  = threadIdx.x;
  const int lane = tid & 31;
  const int w    = tid >> 5;
  const int lh   = lane >> 4;
  const int lr   = lane & 15;

  const int nt      = blockIdx.x % 13;       // query tile
  const int bh      = blockIdx.x / 13;       // 0..639
  const int h       = bh % 8;
  const int bn      = bh / 8;                // b*10+n
  const int rowbase = bn * 200;
  const int qi0     = nt * 16;
  const int valid   = (200 - qi0 < 16) ? (200 - qi0) : 16;
  const float scale = 0.08838834764831845f;  // 128^-0.5

  // ---- stage Q tile (16x128 bf16) via async copies -------------------------
  for (int i = tid; i < 256; i += 128) {     // 256 u16x8 vectors
    const int r = i >> 4, v = i & 15;
    int gr = qi0 + r; if (gr > 199) gr = 199;
    cp_b128(&Qs[r][v * 8],
            Q + (size_t)(rowbase + gr) * 1024 + h * 128 + v * 8);
  }
  cp_wait();
  __syncthreads();

  // ---- scores = Q K^T * scale (WMMA, B frags straight from global K) ------
  for (int jt = w; jt < 13; jt += 4) {
    f32x8 sc = (f32x8)0.0f;
    int jr = jt * 16 + lr; if (jr > 199) jr = 199;
    const unsigned short* kptr = KV + (size_t)(rowbase + jr) * 2048 + h * 128;
#pragma unroll
    for (int kk = 0; kk < 128; kk += 32) {
      const unsigned short* pa = &Qs[lr][kk + lh * 8];
      bf16x16 af = pack_bf16(*(const u16x8*)pa, *(const u16x8*)(pa + 16));
      const unsigned short* pb = kptr + kk + lh * 8;
      bf16x16 bfrag = pack_bf16(*(const u16x8*)pb, *(const u16x8*)(pb + 16));
      sc = __builtin_amdgcn_wmma_f32_16x16x32_bf16(false, af, false, bfrag,
                                                   (short)0, sc, false, false);
    }
#pragma unroll
    for (int r = 0; r < 8; ++r) Ss[lh * 8 + r][jt * 16 + lr] = sc[r] * scale;
  }
  __syncthreads();

  // ---- mask padded columns, add relative-position term --------------------
  for (int i = tid; i < 16 * 24; i += 128) {   // cols 200..223, 16 rows
    const int r = i / 24, cc = 200 + (i % 24);
    Ss[r][cc] = -1e30f;
  }
  for (int p = tid; p < valid * 200; p += 128) {
    const int i = p / 200, j = p % 200;
    const float* rp = rel + (size_t)dist[(qi0 + i) * 200 + j] * 128;
    float s = 0.f;
#pragma unroll 8
    for (int dd = 0; dd < 128; ++dd) s += bf2f(Qs[i][dd]) * rp[dd];
    Ss[i][j] += s * scale;
  }
  __syncthreads();

  // ---- softmax over j (each wave owns 4 rows) ------------------------------
  for (int r = w; r < 16; r += 4) {
    float mx = -1e30f;
    for (int j = lane; j < 208; j += 32) mx = fmaxf(mx, Ss[r][j]);
#pragma unroll
    for (int m = 16; m >= 1; m >>= 1) mx = fmaxf(mx, __shfl_xor(mx, m, 32));
    float sum = 0.f;
    for (int j = lane; j < 224; j += 32) {
      float e = (j < 208) ? __expf(Ss[r][j] - mx) : 0.f;
      Ss[r][j] = e; sum += e;
    }
#pragma unroll
    for (int m = 16; m >= 1; m >>= 1) sum += __shfl_xor(sum, m, 32);
    const float inv = 1.f / sum;
    for (int j = lane; j < 224; j += 32) Ss[r][j] *= inv;
  }
  __syncthreads();

  // ---- out = P V (WMMA), two 128-col V passes staged transposed in LDS ----
  f32x8 oacc[2];
  oacc[0] = (f32x8)0.0f; oacc[1] = (f32x8)0.0f;
  for (int p = 0; p < 2; ++p) {
    const int dd = tid;                 // one dim column per thread
    for (int jj = 0; jj < 128; ++jj) {
      const int j = p * 128 + jj;
      unsigned short v = 0;
      if (j < 200) v = KV[(size_t)(rowbase + j) * 2048 + 1024 + h * 128 + dd];
      Vt[dd][jj] = v;
    }
    __syncthreads();
    const int nsteps = (p == 0) ? 4 : 3;   // p=1: probs beyond 223 are zero
    for (int s = 0; s < nsteps; ++s) {
      const int j0 = p * 128 + s * 32;     // prob col base
      const int jl = s * 32;               // Vt col base
      u16x16 ta;
      const float* ps = &Ss[lr][j0];
#pragma unroll
      for (int j = 0; j < 8; ++j) ta[j]     = f2bf(ps[lh * 8 + j]);
#pragma unroll
      for (int j = 0; j < 8; ++j) ta[8 + j] = f2bf(ps[16 + lh * 8 + j]);
      bf16x16 af = __builtin_bit_cast(bf16x16, ta);
#pragma unroll
      for (int t = 0; t < 2; ++t) {
        const unsigned short* pv = &Vt[w * 32 + t * 16 + lr][jl + lh * 8];
        bf16x16 bfrag = pack_bf16(*(const u16x8*)pv, *(const u16x8*)(pv + 16));
        oacc[t] = __builtin_amdgcn_wmma_f32_16x16x32_bf16(false, af, false, bfrag,
                                                          (short)0, oacc[t], false, false);
      }
    }
    __syncthreads();
  }
  // ---- store concat-head output -------------------------------------------
#pragma unroll
  for (int t = 0; t < 2; ++t) {
    const int dd = w * 32 + t * 16 + lr;
#pragma unroll
    for (int r = 0; r < 8; ++r) {
      const int i = lh * 8 + r;
      if (i < valid)
        O[(size_t)(rowbase + qi0 + i) * 1024 + h * 128 + dd] = f2bf(oacc[t][r]);
    }
  }
}

// ---------------------------------------------------------------------------
// GLU: in [rows,4096] bf16 -> out [rows,2048] bf16 : a * sigmoid(gate)
// ---------------------------------------------------------------------------
__global__ void __launch_bounds__(256) glu_kernel(
    const unsigned short* __restrict__ in, unsigned short* __restrict__ out) {
  const int idx = blockIdx.x * 256 + threadIdx.x;   // rows*2048
  const int c = idx & 2047;
  const int r = idx >> 11;
  const float a = bf2f(in[(size_t)r * 4096 + c]);
  const float g = bf2f(in[(size_t)r * 4096 + 2048 + c]);
  out[idx] = f2bf(a / (1.f + __expf(-g)));
}

// ---------------------------------------------------------------------------
// Depthwise conv (K=15, same-pad) + BN(eval) + SiLU, layout [b, L, C]
// ---------------------------------------------------------------------------
__global__ void __launch_bounds__(256) dwconv_kernel(
    const unsigned short* __restrict__ in, const float* __restrict__ wgt,
    const float* __restrict__ g, const float* __restrict__ b,
    const float* __restrict__ mean, const float* __restrict__ var,
    unsigned short* __restrict__ out) {
  const int idx = blockIdx.x * 256 + threadIdx.x;   // 16000*2048
  const int c  = idx & 2047;
  const int bl = idx >> 11;        // b*2000 + l
  const int l  = bl % 2000;
  float acc = 0.f;
#pragma unroll
  for (int k = 0; k < 15; ++k) {
    const int ll = l + k - 7;
    if (ll >= 0 && ll < 2000)
      acc += bf2f(in[(size_t)(bl + k - 7) * 2048 + c]) * wgt[c * 15 + k];
  }
  float h = (acc - mean[c]) * rsqrtf(var[c] + 1e-5f) * g[c] + b[c];
  out[idx] = f2bf(siluf(h));
}

// ---------------------------------------------------------------------------
// Orchestration
// ---------------------------------------------------------------------------
extern "C" void kernel_launch(void* const* d_in, const int* in_sizes, int n_in,
                              void* d_out, int out_size, void* d_ws, size_t ws_size,
                              hipStream_t stream) {
  (void)in_sizes; (void)n_in; (void)out_size; (void)ws_size;
  const int M = 16000, D = 1024, DFF = 4096, INNER = 2048;

  const float* x          = (const float*)d_in[0];
  const int*   dist       = (const int*)  d_in[1];
  const float* ff1_ln_g   = (const float*)d_in[2];
  const float* ff1_ln_b   = (const float*)d_in[3];
  const float* ff1_up_w   = (const float*)d_in[4];
  const float* ff1_up_b   = (const float*)d_in[5];
  const float* ff1_down_w = (const float*)d_in[6];
  const float* ff1_down_b = (const float*)d_in[7];
  const float* attn_ln_g  = (const float*)d_in[8];
  const float* attn_ln_b  = (const float*)d_in[9];
  const float* wq         = (const float*)d_in[10];
  const float* wkv        = (const float*)d_in[11];
  const float* wo         = (const float*)d_in[12];
  const float* wo_b       = (const float*)d_in[13];
  const float* rel_emb    = (const float*)d_in[14];
  const float* conv_ln_g  = (const float*)d_in[15];
  const float* conv_ln_b  = (const float*)d_in[16];
  const float* conv_up_w  = (const float*)d_in[17];
  const float* conv_up_b  = (const float*)d_in[18];
  const float* dw_w       = (const float*)d_in[19];
  const float* bn_g       = (const float*)d_in[20];
  const float* bn_b       = (const float*)d_in[21];
  const float* bn_mean    = (const float*)d_in[22];
  const float* bn_var     = (const float*)d_in[23];
  const float* conv_down_w= (const float*)d_in[24];
  const float* conv_down_b= (const float*)d_in[25];
  const float* ff2_ln_g   = (const float*)d_in[26];
  const float* ff2_ln_b   = (const float*)d_in[27];
  const float* ff2_up_w   = (const float*)d_in[28];
  const float* ff2_up_b   = (const float*)d_in[29];
  const float* ff2_down_w = (const float*)d_in[30];
  const float* ff2_down_b = (const float*)d_in[31];
  const float* post_ln_g  = (const float*)d_in[32];
  const float* post_ln_b  = (const float*)d_in[33];

  // workspace carve-up (all region sizes are multiples of 256B)
  char* p = (char*)d_ws;
  float*          r0   = (float*)p;          p += (size_t)M * D * 4;      // 64MB
  float*          r1   = (float*)p;          p += (size_t)M * D * 4;
  unsigned short* hbf  = (unsigned short*)p; p += (size_t)M * D * 2;      // LN out
  unsigned short* big  = (unsigned short*)p; p += (size_t)M * DFF * 2;    // ff hidden
  unsigned short* kvbf = (unsigned short*)p; p += (size_t)M * 2 * D * 2;  // kv
  unsigned short* qbf  = (unsigned short*)p; p += (size_t)M * D * 2;      // q
  unsigned short* abf  = (unsigned short*)p; p += (size_t)M * D * 2;      // attn out
  unsigned short* gbf  = kvbf;   // GLU out reuses kv region    [M, INNER]
  unsigned short* cbf  = qbf;    // conv out reuses q+abf span  [M, INNER]

  const dim3 blk(256);
  const dim3 gD (M / 128, D / 128);      // N=1024
  const dim3 gKV(M / 128, 2 * D / 128);  // N=2048
  const dim3 gF (M / 128, DFF / 128);    // N=4096

  // ---- FF1: x + 0.5*FFN(LN(x)) ----
  ln_kernel<0><<<M / 8, blk, 0, stream>>>(x, ff1_ln_g, ff1_ln_b, hbf, nullptr, M, D);
  gemm_bf16<1><<<gF, blk, 0, stream>>>(hbf, ff1_up_w, ff1_up_b, nullptr, nullptr, big, M, DFF, D);
  gemm_bf16<2><<<gD, blk, 0, stream>>>(big, ff1_down_w, ff1_down_b, x, r0, nullptr, M, D, DFF);

  // ---- Attention: r0 + MHA(LN(r0)) ----
  ln_kernel<0><<<M / 8, blk, 0, stream>>>(r0, attn_ln_g, attn_ln_b, hbf, nullptr, M, D);
  gemm_bf16<0><<<gD,  blk, 0, stream>>>(hbf, wq,  nullptr, nullptr, nullptr, qbf,  M, D,     D);
  gemm_bf16<0><<<gKV, blk, 0, stream>>>(hbf, wkv, nullptr, nullptr, nullptr, kvbf, M, 2 * D, D);
  attn_kernel<<<640 * 13, 128, 0, stream>>>(qbf, kvbf, dist, rel_emb, abf);
  gemm_bf16<3><<<gD, blk, 0, stream>>>(abf, wo, wo_b, r0, r1, nullptr, M, D, D);

  // ---- Conv module: r1 + Conv(LN(r1)) ----
  ln_kernel<0><<<M / 8, blk, 0, stream>>>(r1, conv_ln_g, conv_ln_b, hbf, nullptr, M, D);
  gemm_bf16<0><<<gF, blk, 0, stream>>>(hbf, conv_up_w, conv_up_b, nullptr, nullptr, big, M, DFF, D);
  glu_kernel<<<(size_t)M * INNER / 256, blk, 0, stream>>>(big, gbf);
  dwconv_kernel<<<(size_t)M * INNER / 256, blk, 0, stream>>>(gbf, dw_w, bn_g, bn_b, bn_mean, bn_var, cbf);
  gemm_bf16<3><<<gD, blk, 0, stream>>>(cbf, conv_down_w, conv_down_b, r1, r0, nullptr, M, D, INNER);

  // ---- FF2: r0 + 0.5*FFN(LN(r0)) ----
  ln_kernel<0><<<M / 8, blk, 0, stream>>>(r0, ff2_ln_g, ff2_ln_b, hbf, nullptr, M, D);
  gemm_bf16<1><<<gF, blk, 0, stream>>>(hbf, ff2_up_w, ff2_up_b, nullptr, nullptr, big, M, DFF, D);
  gemm_bf16<2><<<gD, blk, 0, stream>>>(big, ff2_down_w, ff2_down_b, r0, r1, nullptr, M, D, DFF);

  // ---- Final LayerNorm -> fp32 output ----
  ln_kernel<1><<<M / 8, blk, 0, stream>>>(r1, post_ln_g, post_ln_b, nullptr, (float*)d_out, M, D);
}
